// ProteinModel_55216099557472
// MI455X (gfx1250) — matrix operand order repaired and checked by reading, not compiled, not run on hardware
//
#include <hip/hip_runtime.h>

typedef unsigned short u16;
typedef unsigned int   u32;

typedef __attribute__((ext_vector_type(16))) __bf16 bf16x16;
typedef __attribute__((ext_vector_type(8)))  float  f32x8;

union FragB16 { bf16x16 v; u32 u[8]; };

#define BATCH   4
#define SEQ     2048
#define BL      8192
#define DM      1280
#define DI      2560
#define DXBC    2816
#define DPROJ   5416
#define DPROJP  5504     // ceil128(5416)
#define NH      40
#define HD      64
#define DSTATE  128
#define DFF     512
#define MHD     160      // 1280 / 8 heads

// ---------------- device helpers ----------------

__device__ __forceinline__ u16 f2bf(float f) {
  u32 u = __float_as_uint(f);
  u32 r = u + 0x7FFFu + ((u >> 16) & 1u);
  return (u16)(r >> 16);
}
__device__ __forceinline__ float siluf(float x) { return x / (1.f + __expf(-x)); }

__device__ __forceinline__ f32x8 zero8() {
  f32x8 z;
  #pragma unroll
  for (int i = 0; i < 8; ++i) z[i] = 0.f;
  return z;
}

__device__ __forceinline__ float block_sum256(float v, float* sh) {
  __syncthreads();
  #pragma unroll
  for (int m = 16; m >= 1; m >>= 1) v += __shfl_xor(v, m, 32);
  const int lane = threadIdx.x & 31, wid = threadIdx.x >> 5;
  if (lane == 0) sh[wid] = v;
  __syncthreads();
  float r = sh[0];
  #pragma unroll
  for (int w = 1; w < 8; ++w) r += sh[w];
  return r;
}

// CDNA5 async global->LDS (ASYNCcnt-tracked); VDST = LDS byte offset VGPR.
__device__ __forceinline__ void async_b128(u32 ldsOff, const u16* g) {
  asm volatile("global_load_async_to_lds_b128 %0, %1, off"
               :: "v"(ldsOff), "v"(g) : "memory");
}
__device__ __forceinline__ void s_wait_async0() { asm volatile("s_wait_asynccnt 0" ::: "memory"); }
__device__ __forceinline__ void s_wait_async4() { asm volatile("s_wait_asynccnt 4" ::: "memory"); }

__device__ __forceinline__ u32 lds_off(const void* p) {
  return (u32)(uintptr_t)p;   // low 32 bits of generic LDS pointer = wave-relative LDS addr
}

// ---------------- converters ----------------

__global__ void cvt_act_bf16(const float* __restrict__ in, u16* __restrict__ out, size_t n) {
  size_t i = (size_t)blockIdx.x * 256 + threadIdx.x;
  size_t stride = (size_t)gridDim.x * 256;
  for (; i < n; i += stride) out[i] = f2bf(in[i]);
}

// W [N,K] f32 -> [Npad,K] bf16, zero rows n>=N
__global__ void cvt_weight_pad(const float* __restrict__ W, u16* __restrict__ Wb,
                               int N, int K, int Npad) {
  size_t i = (size_t)blockIdx.x * 256 + threadIdx.x;
  size_t total = (size_t)Npad * K;
  if (i >= total) return;
  int n = (int)(i / K);
  Wb[i] = (n < N) ? f2bf(W[i]) : (u16)0;
}

// ---------------- WMMA bf16 GEMM:  C[M,N] = A[M,K] @ W[N,K]^T ----------------
// 256 thr = 8 waves; block tile 128x128, BK=32; wave tile 32x64 (2x4 wmma frags).

template<bool BIAS, bool RELU, bool WF32, bool WBF16>
__global__ __launch_bounds__(256)
void gemm_bf16(const u16* __restrict__ A, const u16* __restrict__ B,
               const float* __restrict__ bias,
               float* __restrict__ Cf, u16* __restrict__ Cb,
               int M, int N, int K) {
  __shared__ u16 As[2][128 * 32];
  __shared__ u16 Bs[2][128 * 32];
  const int tid  = threadIdx.x;
  const int bm   = blockIdx.x * 128;
  const int bn   = blockIdx.y * 128;
  const int lane = tid & 31, wid = tid >> 5;
  const int half = lane >> 4, l16 = lane & 15;
  const int mW = (wid >> 1) * 32;
  const int nW = (wid & 1) * 64;
  const int lr = tid >> 2;          // 0..63
  const int lc = (tid & 3) * 8;     // element col (x8 bf16 = 16B)
  const int nk = K / 32;

  f32x8 acc[2][4];
  #pragma unroll
  for (int i = 0; i < 2; ++i)
    #pragma unroll
    for (int j = 0; j < 4; ++j) acc[i][j] = zero8();

  auto issue = [&](int ks, int buf) {
    const u16* aG = A + (size_t)(bm + lr) * K + ks * 32 + lc;
    const u16* bG = B + (size_t)(bn + lr) * K + ks * 32 + lc;
    u32 ao = lds_off(&As[buf][lr * 32 + lc]);
    u32 bo = lds_off(&Bs[buf][lr * 32 + lc]);
    async_b128(ao,            aG);
    async_b128(ao + 64*32*2,  aG + (size_t)64 * K);
    async_b128(bo,            bG);
    async_b128(bo + 64*32*2,  bG + (size_t)64 * K);
  };

  issue(0, 0);
  for (int ks = 0; ks < nk; ++ks) {
    const int buf = ks & 1;
    if (ks + 1 < nk) { issue(ks + 1, buf ^ 1); s_wait_async4(); }
    else             { s_wait_async0(); }
    __syncthreads();

    const u16* as = As[buf];
    const u16* bs = Bs[buf];
    FragB16 af[2], bfr[4];
    #pragma unroll
    for (int i = 0; i < 2; ++i) {
      int m = mW + i * 16 + l16;
      #pragma unroll
      for (int v = 0; v < 8; ++v) {
        int kp = 2 * v + 16 * half;                          // A 16x32 layout
        af[i].u[v] = *(const u32*)&as[m * 32 + kp];
      }
    }
    #pragma unroll
    for (int j = 0; j < 4; ++j) {
      int n = nW + j * 16 + l16;
      #pragma unroll
      for (int v = 0; v < 8; ++v) {
        int kp = 2 * v + 16 * half;                          // B 32x16 layout
        bfr[j].u[v] = *(const u32*)&bs[n * 32 + kp];
      }
    }
    #pragma unroll
    for (int i = 0; i < 2; ++i)
      #pragma unroll
      for (int j = 0; j < 4; ++j)
        acc[i][j] = __builtin_amdgcn_wmma_f32_16x16x32_bf16(
            false, af[i].v, false, bfr[j].v, (short)0, acc[i][j], false, false);

    __syncthreads();
  }

  #pragma unroll
  for (int i = 0; i < 2; ++i)
    #pragma unroll
    for (int j = 0; j < 4; ++j)
      #pragma unroll
      for (int r = 0; r < 8; ++r) {
        int row = bm + mW + i * 16 + r + 8 * half;           // D layout
        int col = bn + nW + j * 16 + l16;
        if (col < N) {
          float v = acc[i][j][r];
          if (BIAS) v += bias[col];
          if (RELU) v = fmaxf(v, 0.f);
          if (WF32)  Cf[(size_t)row * N + col] = v;
          if (WBF16) Cb[(size_t)row * N + col] = f2bf(v);
        }
      }
}

// ---------------- flash attention (bf16 in, bf16 out, f32 accum) ----------------
// grid (qt=16, head=8, batch=4); 256 thr; wave owns 16 query rows; key tile 64.

__global__ __launch_bounds__(256)
void flash_attn(const u16* __restrict__ qkv, u16* __restrict__ out) {
  __shared__ u16 Ks[64 * MHD];
  __shared__ u16 Vs[64 * MHD];
  __shared__ u16 Ps[8][16 * 64];
  const int qt = blockIdx.x, h = blockIdx.y, b = blockIdx.z;
  const int tid = threadIdx.x, lane = tid & 31, wid = tid >> 5;
  const int half = lane >> 4, l16 = lane & 15;
  const size_t rs = 3 * DM;
  const size_t base = (size_t)b * SEQ * rs;
  const int qRow0 = qt * 128 + wid * 16;

  FragB16 qf[5];
  {
    const u16* qp = qkv + base + (size_t)(qRow0 + l16) * rs + h * MHD;
    #pragma unroll
    for (int f = 0; f < 5; ++f)
      #pragma unroll
      for (int v = 0; v < 8; ++v) {
        int kp = f * 32 + 2 * v + 16 * half;
        qf[f].u[v] = *(const u32*)&qp[kp];
      }
  }
  float mrun[8], lrun[8];
  #pragma unroll
  for (int r = 0; r < 8; ++r) { mrun[r] = -1e30f; lrun[r] = 0.f; }
  f32x8 o[10];
  #pragma unroll
  for (int n = 0; n < 10; ++n) o[n] = zero8();
  const float scale = 0.07905694150420949f;   // 1/sqrt(160)

  for (int t = 0; t < SEQ / 64; ++t) {
    const int kt0 = t * 64;
    #pragma unroll
    for (int i = 0; i < 5; ++i) {
      int chunk = tid + i * 256;               // 1280 chunks of 16B per matrix
      int row = chunk / 20, c = (chunk % 20) * 8;
      const u16* kg = qkv + base + (size_t)(kt0 + row) * rs + DM     + h * MHD + c;
      const u16* vg = qkv + base + (size_t)(kt0 + row) * rs + 2 * DM + h * MHD + c;
      async_b128(lds_off(&Ks[row * MHD + c]), kg);
      async_b128(lds_off(&Vs[row * MHD + c]), vg);
    }
    s_wait_async0();
    __syncthreads();

    // S = Q K^T
    f32x8 s[4];
    #pragma unroll
    for (int j = 0; j < 4; ++j) s[j] = zero8();
    #pragma unroll
    for (int kf = 0; kf < 5; ++kf)
      #pragma unroll
      for (int j = 0; j < 4; ++j) {
        FragB16 bf;
        int n = j * 16 + l16;
        #pragma unroll
        for (int v = 0; v < 8; ++v) {
          int kp = kf * 32 + 2 * v + 16 * half;
          bf.u[v] = *(const u32*)&Ks[n * MHD + kp];
        }
        s[j] = __builtin_amdgcn_wmma_f32_16x16x32_bf16(
            false, qf[kf].v, false, bf.v, (short)0, s[j], false, false);
      }

    // online softmax (row reductions across 16-lane halves)
    float alpha[8];
    #pragma unroll
    for (int r = 0; r < 8; ++r) {
      float mt = -1e30f;
      #pragma unroll
      for (int j = 0; j < 4; ++j) mt = fmaxf(mt, s[j][r] * scale);
      #pragma unroll
      for (int m2 = 1; m2 < 16; m2 <<= 1) mt = fmaxf(mt, __shfl_xor(mt, m2, 32));
      float mn = fmaxf(mrun[r], mt);
      alpha[r] = __expf(mrun[r] - mn);
      float lt = 0.f;
      #pragma unroll
      for (int j = 0; j < 4; ++j) {
        float p = __expf(s[j][r] * scale - mn);
        s[j][r] = p;
        lt += p;
      }
      #pragma unroll
      for (int m2 = 1; m2 < 16; m2 <<= 1) lt += __shfl_xor(lt, m2, 32);
      lrun[r] = lrun[r] * alpha[r] + lt;
      mrun[r] = mn;
    }
    #pragma unroll
    for (int n = 0; n < 10; ++n)
      #pragma unroll
      for (int r = 0; r < 8; ++r) o[n][r] *= alpha[r];

    // P -> per-wave LDS (re-fragment D layout -> A layout)
    #pragma unroll
    for (int j = 0; j < 4; ++j)
      #pragma unroll
      for (int r = 0; r < 8; ++r)
        Ps[wid][(r + 8 * half) * 64 + j * 16 + l16] = f2bf(s[j][r]);

    // O += P @ V
    #pragma unroll
    for (int kf = 0; kf < 2; ++kf) {
      FragB16 af;
      #pragma unroll
      for (int v = 0; v < 8; ++v) {
        int kp = kf * 32 + 2 * v + 16 * half;
        af.u[v] = *(const u32*)&Ps[wid][l16 * 64 + kp];
      }
      #pragma unroll
      for (int n = 0; n < 10; ++n) {
        FragB16 bf;
        int nn = n * 16 + l16;
        #pragma unroll
        for (int v = 0; v < 8; ++v) {
          int k = kf * 32 + 2 * v + 16 * half;
          bf.u[v] = (u32)Vs[k * MHD + nn] | ((u32)Vs[(k + 1) * MHD + nn] << 16);
        }
        o[n] = __builtin_amdgcn_wmma_f32_16x16x32_bf16(
            false, af.v, false, bf.v, (short)0, o[n], false, false);
      }
    }
    __syncthreads();
  }

  #pragma unroll
  for (int r = 0; r < 8; ++r) {
    float inv = 1.f / lrun[r];
    int row = qRow0 + r + 8 * half;
    size_t orow = ((size_t)b * SEQ + row) * DM + h * MHD;
    #pragma unroll
    for (int n = 0; n < 10; ++n)
      out[orow + n * 16 + l16] = f2bf(o[n][r] * inv);
  }
}

// ---------------- mamba pieces ----------------

__global__ void conv_silu(const float* __restrict__ zx, const float* __restrict__ cw,
                          const float* __restrict__ cb, float* __restrict__ xc) {
  const int c = blockIdx.y * 256 + threadIdx.x;
  if (c >= DXBC) return;
  const int row = blockIdx.x;
  const int l = row & (SEQ - 1);
  float acc = cb[c];
  #pragma unroll
  for (int k = 0; k < 4; ++k) {
    int lk = l - 3 + k;
    if (lk >= 0)
      acc += zx[(size_t)(row - 3 + k) * DPROJ + DI + c] * cw[c * 4 + k];
  }
  xc[(size_t)row * DXBC + c] = siluf(acc);
}

__global__ void dt_kernel(const float* __restrict__ zx, const float* __restrict__ dtb,
                          const float* __restrict__ Alog,
                          float* __restrict__ dts, float* __restrict__ dAa) {
  int i = blockIdx.x * 256 + threadIdx.x;
  if (i >= BL * NH) return;
  int row = i / NH, h = i % NH;
  float xv = zx[(size_t)row * DPROJ + (DI + DXBC) + h] + dtb[h];
  float sp = (xv > 20.f) ? xv : log1pf(__expf(xv));
  dts[i] = sp;
  dAa[i] = __expf(-__expf(Alog[h]) * sp);
}

__global__ __launch_bounds__(256)
void mamba_scan(const float* __restrict__ xc, const float* __restrict__ dts,
                const float* __restrict__ dAa, const float* __restrict__ Dp,
                float* __restrict__ y) {
  __shared__ float Bt[DSTATE], Ct[DSTATE];
  const int bh = blockIdx.x;
  const int b = bh / NH, h = bh % NH;
  const int tid = threadIdx.x;
  const int p  = tid >> 2;
  const int nb = (tid & 3) * 32;
  float st[32];
  #pragma unroll
  for (int j = 0; j < 32; ++j) st[j] = 0.f;
  const float Dv = Dp[h];
  for (int t = 0; t < SEQ; ++t) {
    const size_t row = (size_t)b * SEQ + t;
    if (tid < DSTATE) Bt[tid] = xc[row * DXBC + DI + tid];
    else              Ct[tid - DSTATE] = xc[row * DXBC + DI + DSTATE + (tid - DSTATE)];
    __syncthreads();
    const float xv = xc[row * DXBC + h * HD + p];
    const size_t ih = row * NH + h;
    const float a  = dAa[ih];
    const float co = dts[ih] * xv;
    float acc = 0.f;
    #pragma unroll
    for (int j = 0; j < 32; ++j) {
      st[j] = st[j] * a + co * Bt[nb + j];
      acc += st[j] * Ct[nb + j];
    }
    acc += __shfl_xor(acc, 1, 32);
    acc += __shfl_xor(acc, 2, 32);
    if ((tid & 3) == 0) y[row * DI + h * HD + p] = acc + Dv * xv;
    __syncthreads();
  }
}

__global__ __launch_bounds__(256)
void gated_rmsnorm(const float* __restrict__ y, const float* __restrict__ zx,
                   const float* __restrict__ nw, u16* __restrict__ yb) {
  __shared__ float red[8];
  const size_t row = blockIdx.x;
  float g[10]; float ss = 0.f;
  #pragma unroll
  for (int i = 0; i < 10; ++i) {
    int c = threadIdx.x + i * 256;
    float z = zx[row * DPROJ + c];
    float gv = y[row * DI + c] * siluf(z);
    g[i] = gv; ss += gv * gv;
  }
  float s = block_sum256(ss, red);
  float r = rsqrtf(s / DI + 1e-5f);
  #pragma unroll
  for (int i = 0; i < 10; ++i) {
    int c = threadIdx.x + i * 256;
    yb[row * DI + c] = f2bf(g[i] * r * nw[c]);
  }
}

__global__ __launch_bounds__(256)
void ln_residual(const float* __restrict__ X, const float* __restrict__ R,
                 const float* __restrict__ g, const float* __restrict__ bta,
                 float* __restrict__ Of, u16* __restrict__ Ob) {
  __shared__ float red[8];
  const size_t row = blockIdx.x;
  float v[5]; float s = 0.f;
  #pragma unroll
  for (int i = 0; i < 5; ++i) {
    int c = threadIdx.x + i * 256;
    float val = X[row * DM + c] + R[row * DM + c];
    v[i] = val; s += val;
  }
  float mean = block_sum256(s, red) / DM;
  float s2 = 0.f;
  #pragma unroll
  for (int i = 0; i < 5; ++i) { float d = v[i] - mean; s2 += d * d; }
  float var = block_sum256(s2, red) / DM;
  float rsv = rsqrtf(var + 1e-5f);
  #pragma unroll
  for (int i = 0; i < 5; ++i) {
    int c = threadIdx.x + i * 256;
    float o = (v[i] - mean) * rsv * g[c] + bta[c];
    Of[row * DM + c] = o;
    Ob[row * DM + c] = f2bf(o);
  }
}

// ---------------- heads + CRF ----------------

__global__ __launch_bounds__(256)
void em_head(const float* __restrict__ e, const float* __restrict__ lnw,
             const float* __restrict__ lnb, const float* __restrict__ Wt2,
             const float* __restrict__ bt2, float* __restrict__ em) {
  __shared__ float red[8];
  const int t = blockIdx.x, b = blockIdx.y, tid = threadIdx.x;
  const size_t row = (size_t)b * SEQ + 1 + t;
  float v = e[row * 256 + tid];                 // relu already applied in GEMM
  float s1 = block_sum256(v, red);
  float s2 = block_sum256(v * v, red);
  float mean = s1 / 256.f, var = s2 / 256.f - mean * mean;
  float cn = (v - mean) * rsqrtf(var + 1e-5f) * lnw[tid] + lnb[tid];
  float p0 = block_sum256(cn * Wt2[tid], red);
  float p1 = block_sum256(cn * Wt2[256 + tid], red);
  if (tid == 0) {
    size_t o = ((size_t)b * (SEQ - 2) + t) * 2;
    em[o] = p0 + bt2[0];
    em[o + 1] = p1 + bt2[1];
  }
}

__device__ __forceinline__ float lse2(float a, float b) {
  float m = fmaxf(a, b);
  return m + logf(__expf(a - m) + __expf(b - m));
}

__global__ void crf_kernel(const float* __restrict__ em, const int* __restrict__ tags,
                           const float* __restrict__ trans, const float* __restrict__ st,
                           const float* __restrict__ et, float* __restrict__ out) {
  const int b = threadIdx.x;
  float res = 0.f;
  if (b < BATCH) {
    const float T00 = trans[0], T01 = trans[1], T10 = trans[2], T11 = trans[3];
    const float* e = em + (size_t)b * (SEQ - 2) * 2;
    const int* tg = tags + (size_t)b * (SEQ - 2);
    int prev = tg[0];
    float score = st[prev] + e[prev];
    float a0 = st[0] + e[0], a1 = st[1] + e[1];
    for (int t = 1; t < SEQ - 2; ++t) {
      float e0 = e[t * 2], e1 = e[t * 2 + 1];
      int ct = tg[t];
      score += (ct == 0 ? e0 : e1) + trans[prev * 2 + ct];
      prev = ct;
      float n0 = lse2(a0 + T00, a1 + T10) + e0;
      float n1 = lse2(a0 + T01, a1 + T11) + e1;
      a0 = n0; a1 = n1;
    }
    score += et[prev];
    res = score - lse2(a0 + et[0], a1 + et[1]);
  }
  res += __shfl_xor(res, 1, 32);
  res += __shfl_xor(res, 2, 32);
  if (threadIdx.x == 0) out[0] = -res * 0.25f;
}

__global__ __launch_bounds__(256)
void class_head(const float* __restrict__ h3, const float* __restrict__ Wc1,
                const float* __restrict__ bc1, const float* __restrict__ lnw,
                const float* __restrict__ lnb, const float* __restrict__ Wc2,
                const float* __restrict__ bc2, float* __restrict__ out) {
  __shared__ float hrow[DM];
  __shared__ float carr[256];
  __shared__ float red[8];
  const int b = blockIdx.x, tid = threadIdx.x;
  const size_t r0 = (size_t)b * SEQ * DM;
  #pragma unroll
  for (int i = 0; i < 5; ++i) hrow[tid + i * 256] = h3[r0 + tid + i * 256];
  __syncthreads();
  float acc = bc1[tid];
  for (int k = 0; k < DM; ++k) acc += hrow[k] * Wc1[(size_t)tid * DM + k];
  acc = fmaxf(acc, 0.f);
  float s1 = block_sum256(acc, red);
  float s2 = block_sum256(acc * acc, red);
  float mean = s1 / 256.f, var = s2 / 256.f - mean * mean;
  float cn = (acc - mean) * rsqrtf(var + 1e-5f) * lnw[tid] + lnb[tid];
  carr[tid] = cn;
  __syncthreads();
  if (tid < 5) {
    float o = bc2[tid];
    for (int k = 0; k < 256; ++k) o += carr[k] * Wc2[tid * 256 + k];
    out[b * 5 + tid] = o;
  }
}

// ---------------- launcher ----------------

extern "C" void kernel_launch(void* const* d_in, const int* in_sizes, int n_in,
                              void* d_out, int out_size, void* d_ws, size_t ws_size,
                              hipStream_t stream) {
  (void)in_sizes; (void)n_in; (void)out_size; (void)ws_size;
  const float* x       = (const float*)d_in[0];
  const int*   tags    = (const int*)  d_in[1];
  const float* W_in    = (const float*)d_in[2];
  const float* conv_w  = (const float*)d_in[3];
  const float* conv_b  = (const float*)d_in[4];
  const float* dt_bias = (const float*)d_in[5];
  const float* A_log   = (const float*)d_in[6];
  const float* D_par   = (const float*)d_in[7];
  const float* norm_w  = (const float*)d_in[8];
  const float* W_out   = (const float*)d_in[9];
  const float* ln_m_w  = (const float*)d_in[10];
  const float* ln_m_b  = (const float*)d_in[11];
  const float* W_qkv   = (const float*)d_in[12];
  const float* b_qkv   = (const float*)d_in[13];
  const float* W_o     = (const float*)d_in[14];
  const float* b_o     = (const float*)d_in[15];
  const float* ln_a_w  = (const float*)d_in[16];
  const float* ln_a_b  = (const float*)d_in[17];
  const float* W1      = (const float*)d_in[18];
  const float* b1      = (const float*)d_in[19];
  const float* W2      = (const float*)d_in[20];
  const float* b2      = (const float*)d_in[21];
  const float* ln_f_w  = (const float*)d_in[22];
  const float* ln_f_b  = (const float*)d_in[23];
  const float* Wc1     = (const float*)d_in[24];
  const float* bc1     = (const float*)d_in[25];
  const float* lnc_w   = (const float*)d_in[26];
  const float* lnc_b   = (const float*)d_in[27];
  const float* Wc2     = (const float*)d_in[28];
  const float* bc2     = (const float*)d_in[29];
  const float* Wt1     = (const float*)d_in[30];
  const float* bt1     = (const float*)d_in[31];
  const float* lnt_w   = (const float*)d_in[32];
  const float* lnt_b   = (const float*)d_in[33];
  const float* Wt2     = (const float*)d_in[34];
  const float* bt2     = (const float*)d_in[35];
  const float* trans   = (const float*)d_in[36];
  const float* start_t = (const float*)d_in[37];
  const float* end_t   = (const float*)d_in[38];
  float* out = (float*)d_out;

  char* wp = (char*)d_ws;
  auto alloc = [&](size_t bytes) -> void* {
    void* r = (void*)wp;
    wp += (bytes + 255) & ~(size_t)255;
    return r;
  };
  u16* Win_b  = (u16*)alloc((size_t)DPROJP * DM * 2);
  u16* Wout_b = (u16*)alloc((size_t)DM * DI * 2);
  u16* Wqkv_b = (u16*)alloc((size_t)3 * DM * DM * 2);
  u16* Wo_b   = (u16*)alloc((size_t)DM * DM * 2);
  u16* W1_b   = (u16*)alloc((size_t)DFF * DM * 2);
  u16* W2_b   = (u16*)alloc((size_t)DM * DFF * 2);
  u16* Wt1_b  = (u16*)alloc((size_t)256 * DM * 2);
  u16* x_b    = (u16*)alloc((size_t)BL * DM * 2);
  float* zx   = (float*)alloc((size_t)BL * DPROJ * 4);
  float* xc   = (float*)alloc((size_t)BL * DXBC * 4);
  float* dts  = (float*)alloc((size_t)BL * NH * 4);
  float* dAa  = (float*)alloc((size_t)BL * NH * 4);
  float* yf   = (float*)alloc((size_t)BL * DI * 4);
  u16*   y_b  = (u16*)alloc((size_t)BL * DI * 2);
  float* mo   = (float*)alloc((size_t)BL * DM * 4);
  float* h1   = (float*)alloc((size_t)BL * DM * 4);
  u16*   h1b  = (u16*)alloc((size_t)BL * DM * 2);
  u16*   qkvb = (u16*)alloc((size_t)BL * 3 * DM * 2);
  u16*   attb = (u16*)alloc((size_t)BL * DM * 2);
  float* oP   = (float*)alloc((size_t)BL * DM * 4);
  float* h2   = (float*)alloc((size_t)BL * DM * 4);
  u16*   h2b  = (u16*)alloc((size_t)BL * DM * 2);
  u16*   ff1b = (u16*)alloc((size_t)BL * DFF * 2);
  float* ff2  = (float*)alloc((size_t)BL * DM * 4);
  float* h3   = (float*)alloc((size_t)BL * DM * 4);
  u16*   h3b  = (u16*)alloc((size_t)BL * DM * 2);
  float* ef   = (float*)alloc((size_t)BL * 256 * 4);
  float* emb  = (float*)alloc((size_t)BATCH * (SEQ - 2) * 2 * 4);

  auto cvtw = [&](const float* W, u16* Wb, int N, int K, int Npad) {
    size_t tot = (size_t)Npad * K;
    cvt_weight_pad<<<(unsigned)((tot + 255) / 256), 256, 0, stream>>>(W, Wb, N, K, Npad);
  };
  cvtw(W_in,  Win_b,  DPROJ,  DM,  DPROJP);
  cvtw(W_out, Wout_b, DM,     DI,  DM);
  cvtw(W_qkv, Wqkv_b, 3 * DM, DM,  3 * DM);
  cvtw(W_o,   Wo_b,   DM,     DM,  DM);
  cvtw(W1,    W1_b,   DFF,    DM,  DFF);
  cvtw(W2,    W2_b,   DM,     DFF, DM);
  cvtw(Wt1,   Wt1_b,  256,    DM,  256);
  cvt_act_bf16<<<4096, 256, 0, stream>>>(x, x_b, (size_t)BL * DM);

  // Mamba2
  gemm_bf16<false, false, true, false><<<dim3(BL / 128, DPROJP / 128), 256, 0, stream>>>(
      x_b, Win_b, nullptr, zx, nullptr, BL, DPROJ, DM);
  conv_silu<<<dim3(BL, (DXBC + 255) / 256), 256, 0, stream>>>(zx, conv_w, conv_b, xc);
  dt_kernel<<<(BL * NH + 255) / 256, 256, 0, stream>>>(zx, dt_bias, A_log, dts, dAa);
  mamba_scan<<<BATCH * NH, 256, 0, stream>>>(xc, dts, dAa, D_par, yf);
  gated_rmsnorm<<<BL, 256, 0, stream>>>(yf, zx, norm_w, y_b);
  gemm_bf16<false, false, true, false><<<dim3(BL / 128, DM / 128), 256, 0, stream>>>(
      y_b, Wout_b, nullptr, mo, nullptr, BL, DM, DI);
  ln_residual<<<BL, 256, 0, stream>>>(x, mo, ln_m_w, ln_m_b, h1, h1b);

  // MHA
  gemm_bf16<true, false, false, true><<<dim3(BL / 128, (3 * DM) / 128), 256, 0, stream>>>(
      h1b, Wqkv_b, b_qkv, nullptr, qkvb, BL, 3 * DM, DM);
  flash_attn<<<dim3(SEQ / 128, 8, BATCH), 256, 0, stream>>>(qkvb, attb);
  gemm_bf16<true, false, true, false><<<dim3(BL / 128, DM / 128), 256, 0, stream>>>(
      attb, Wo_b, b_o, oP, nullptr, BL, DM, DM);
  ln_residual<<<BL, 256, 0, stream>>>(h1, oP, ln_a_w, ln_a_b, h2, h2b);

  // FFN
  gemm_bf16<true, true, false, true><<<dim3(BL / 128, DFF / 128), 256, 0, stream>>>(
      h2b, W1_b, b1, nullptr, ff1b, BL, DFF, DM);
  gemm_bf16<true, false, true, false><<<dim3(BL / 128, DM / 128), 256, 0, stream>>>(
      ff1b, W2_b, b2, ff2, nullptr, BL, DM, DFF);
  ln_residual<<<BL, 256, 0, stream>>>(h2, ff2, ln_f_w, ln_f_b, h3, h3b);

  // heads
  gemm_bf16<true, true, true, false><<<dim3(BL / 128, 256 / 128), 256, 0, stream>>>(
      h3b, Wt1_b, bt1, ef, nullptr, BL, 256, DM);
  em_head<<<dim3(SEQ - 2, BATCH), 256, 0, stream>>>(ef, lnt_w, lnt_b, Wt2, bt2, emb);
  crf_kernel<<<1, 32, 0, stream>>>(emb, tags, trans, start_t, end_t, out + 20);
  class_head<<<BATCH, 256, 0, stream>>>(h3, Wc1, bc1, lnc_w, lnc_b, Wc2, bc2, out);
}
